// Attention_52621939311051
// MI455X (gfx1250) — matrix operand order repaired and checked by reading, not compiled, not run on hardware
//
#include <hip/hip_runtime.h>

#define HIDDEN    1024
#define HEADS     16
#define HEAD_DIM  64
#define SEQ       1024
#define BATCH     8
#define ROWS      (BATCH * SEQ)      // 8192
#define SCALE     0.125f             // 64^-0.5
#define NEG_INF   (-4294967295.0f)   // -(2^32 - 1)

typedef __bf16 bf16;
typedef bf16  bf16x16 __attribute__((ext_vector_type(16)));
typedef float f32x8   __attribute__((ext_vector_type(8)));
typedef unsigned int u32;
typedef u32   u32x4   __attribute__((ext_vector_type(4)));

// ---------------------------------------------------------------- helpers

__device__ __forceinline__ unsigned short f2bf_bits(float f) {
    u32 u = __float_as_uint(f);
    u32 r = u + 0x7FFFu + ((u >> 16) & 1u);   // round-to-nearest-even
    return (unsigned short)(r >> 16);
}
__device__ __forceinline__ bf16 f2bf16(float f) {
    union { unsigned short u; bf16 h; } cv;
    cv.u = f2bf_bits(f);
    return cv.h;
}

union FragU { bf16x16 v; u32x4 q[2]; };
union Chunk8 { u32x4 q; bf16 h[8]; };

// Async DMA: global -> LDS, 16B per lane, tracked by ASYNCcnt (CDNA5).
// GVS addressing: mem = SGPR base + 32-bit VGPR offset; VDST = LDS byte addr.
__device__ __forceinline__ void async_load_b128(unsigned ldsAddr,
                                                const void* gbase,
                                                unsigned gByteOff) {
    asm volatile("global_load_async_to_lds_b128 %0, %1, %2"
                 :: "v"(ldsAddr), "v"(gByteOff),
                    "s"((unsigned long long)(uintptr_t)gbase)
                 : "memory");
}
__device__ __forceinline__ void wait_async0() {
    asm volatile("s_wait_asynccnt 0x0" ::: "memory");
}
__device__ __forceinline__ unsigned lds_addr(const void* p) {
    return (unsigned)(uintptr_t)p;   // low 32 bits of shared-aperture address
}

// A operand 16x32 (MxK), ISA layout: lanes 0-15 row=l K 0..7 & 16..23,
// lanes 16-31 row=l K 8..15 & 24..31. Tile in LDS is [m][k], k contiguous.
__device__ __forceinline__ bf16x16 load_frag_a(const bf16* lds, int rowStride,
                                               int rowBase, int kBase, int lane) {
    int half = lane >> 4, l = lane & 15;
    const bf16* p = lds + (size_t)(rowBase + l) * rowStride + kBase + half * 8;
    FragU f;
    f.q[0] = *(const u32x4*)(p);
    f.q[1] = *(const u32x4*)(p + 16);
    return f.v;
}

// B operand 32x16 (KxN), ISA layout: lanes 0-15 col=l K 0..7 & 16..23,
// lanes 16-31 col=l K 8..15 & 24..31. Tile in LDS is [n][k].
__device__ __forceinline__ bf16x16 load_frag_b(const bf16* lds, int rowStride,
                                               int colBase, int kBase, int lane) {
    int half = lane >> 4, l = lane & 15;
    const bf16* p = lds + (size_t)(colBase + l) * rowStride + kBase + half * 8;
    FragU f;
    f.q[0] = *(const u32x4*)(p);
    f.q[1] = *(const u32x4*)(p + 16);
    return f.v;
}

__device__ __forceinline__ f32x8 wmma_bf16(bf16x16 a, bf16x16 b, f32x8 c) {
    return __builtin_amdgcn_wmma_f32_16x16x32_bf16(
        false, a, false, b, (short)0, c, false, false);
}

// ---------------------------------------------------------------- convert

__global__ __launch_bounds__(256)
void convert_f32_to_bf16_kernel(const float* __restrict__ in,
                                bf16* __restrict__ out, int n4) {
    int i = blockIdx.x * blockDim.x + threadIdx.x;
    if (i >= n4) return;
    float4 v = ((const float4*)in)[i];
    bf16* o = out + (size_t)i * 4;
    o[0] = f2bf16(v.x); o[1] = f2bf16(v.y);
    o[2] = f2bf16(v.z); o[3] = f2bf16(v.w);
}

// W[K][N] f32 -> Wt[N][K] bf16  (32x32 tiles via padded LDS)
__global__ __launch_bounds__(256)
void transpose_convert_kernel(const float* __restrict__ W, bf16* __restrict__ Wt) {
    __shared__ float t[32][33];
    int tx = threadIdx.x & 31, ty = threadIdx.x >> 5;  // ty 0..7
    int k0 = blockIdx.x * 32, n0 = blockIdx.y * 32;
    #pragma unroll
    for (int i = 0; i < 4; ++i) {
        int r = ty + i * 8;
        t[r][tx] = W[(size_t)(k0 + r) * HIDDEN + n0 + tx];
    }
    __syncthreads();
    #pragma unroll
    for (int i = 0; i < 4; ++i) {
        int r = ty + i * 8;
        Wt[(size_t)(n0 + r) * HIDDEN + k0 + tx] = f2bf16(t[tx][r]);
    }
}

// ---------------------------------------------------------------- GEMM
// C[M=8192][N=1024] = A_bf16[M][K] * Wt_bf16[N][K]^T + bias
// Block tile 128x128, BK=64, double-buffered LDS filled by async DMA.
// 256 threads = 8 waves, wave tile 64x32, 16 WMMA per K-step.
// OUT_BHND=1: write bf16 scattered to [B,H,N,D]; OUT_BHND=0: f32 row-major.

template <int OUT_BHND>
__global__ __launch_bounds__(256)
void gemm_bf16_kernel(const bf16* __restrict__ A, const bf16* __restrict__ Bt,
                      const float* __restrict__ bias, void* __restrict__ outPtr) {
    const int K = HIDDEN, N = HIDDEN;
    __shared__ __align__(16) bf16 sA[2][128 * 64];
    __shared__ __align__(16) bf16 sB[2][128 * 64];

    int tid  = threadIdx.x;
    int lane = tid & 31, wave = tid >> 5;
    int waveM = wave >> 2, waveN = wave & 3;            // 2 x 4 wave grid
    int m0 = blockIdx.y * 128, n0 = blockIdx.x * 128;

    // this thread's 4 DMA chunks (row, k-offset within tile)
    int rowC[4], kcC[4];
    unsigned aBase[4], bBase[4];     // global byte offsets at k0 = 0
    #pragma unroll
    for (int i = 0; i < 4; ++i) {
        int c = tid + 256 * i;
        rowC[i] = c >> 3;
        kcC[i]  = (c & 7) * 8;
        aBase[i] = (unsigned)((m0 + rowC[i]) * K + kcC[i]) * 2u;
        bBase[i] = (unsigned)((n0 + rowC[i]) * K + kcC[i]) * 2u;
    }

    // prologue: DMA tile 0 into buffer 0
    #pragma unroll
    for (int i = 0; i < 4; ++i) {
        unsigned dst = (unsigned)(rowC[i] * 64 + kcC[i]) * 2u;
        async_load_b128(lds_addr(&sA[0][0]) + dst, A,  aBase[i]);
        async_load_b128(lds_addr(&sB[0][0]) + dst, Bt, bBase[i]);
    }
    wait_async0();
    __syncthreads();

    f32x8 acc[4][2];
    #pragma unroll
    for (int tm = 0; tm < 4; ++tm)
        #pragma unroll
        for (int tn = 0; tn < 2; ++tn)
            #pragma unroll
            for (int e = 0; e < 8; ++e) acc[tm][tn][e] = 0.0f;

    for (int k0 = 0; k0 < K; k0 += 64) {
        int p = (k0 >> 6) & 1;

        // kick off DMA for the next K-tile into the other buffer
        if (k0 + 64 < K) {
            unsigned gAdd = (unsigned)(k0 + 64) * 2u;
            #pragma unroll
            for (int i = 0; i < 4; ++i) {
                unsigned dst = (unsigned)(rowC[i] * 64 + kcC[i]) * 2u;
                async_load_b128(lds_addr(&sA[p ^ 1][0]) + dst, A,  aBase[i] + gAdd);
                async_load_b128(lds_addr(&sB[p ^ 1][0]) + dst, Bt, bBase[i] + gAdd);
            }
        }

        // matrix math on current buffer (overlaps with DMA above)
        const bf16* cA = &sA[p][0];
        const bf16* cB = &sB[p][0];
        bf16x16 bf[2][2];
        #pragma unroll
        for (int tn = 0; tn < 2; ++tn)
            #pragma unroll
            for (int kk = 0; kk < 2; ++kk)
                bf[tn][kk] = load_frag_b(cB, 64, waveN * 32 + tn * 16, kk * 32, lane);

        #pragma unroll
        for (int tm = 0; tm < 4; ++tm) {
            #pragma unroll
            for (int kk = 0; kk < 2; ++kk) {
                bf16x16 af = load_frag_a(cA, 64, waveM * 64 + tm * 16, kk * 32, lane);
                #pragma unroll
                for (int tn = 0; tn < 2; ++tn)
                    acc[tm][tn] = wmma_bf16(af, bf[tn][kk], acc[tm][tn]);
            }
        }

        wait_async0();       // next tile landed in LDS
        __syncthreads();     // all waves done reading buffer p / writing p^1
    }

    // epilogue: C layout -> VGPR e holds row (e + 8*half), col = lane&15
    int half = lane >> 4, l = lane & 15;
    #pragma unroll
    for (int tm = 0; tm < 4; ++tm) {
        #pragma unroll
        for (int tn = 0; tn < 2; ++tn) {
            int rBase = m0 + waveM * 64 + tm * 16 + half * 8;
            int cIdx  = n0 + waveN * 32 + tn * 16 + l;
            float bv = bias[cIdx];
            #pragma unroll
            for (int e = 0; e < 8; ++e) {
                int   r = rBase + e;
                float v = acc[tm][tn][e] + bv;
                if (OUT_BHND) {
                    int b = r >> 10, n = r & 1023;
                    int h = cIdx >> 6, d = cIdx & 63;
                    ((bf16*)outPtr)[(((size_t)(b * HEADS + h) * SEQ + n) * HEAD_DIM) + d]
                        = f2bf16(v);
                } else {
                    ((float*)outPtr)[(size_t)r * N + cIdx] = v;
                }
            }
        }
    }
}

// ---------------------------------------------------------------- attention
// Q/K/V: [B][H][N][D] bf16.  1 block = 4 waves = 64 query rows of one (b,h).
// Flash-attention with online softmax; out: attnOut[B*N][H*D] bf16.

__global__ __launch_bounds__(128)
void attention_kernel(const bf16* __restrict__ Q, const bf16* __restrict__ Km,
                      const bf16* __restrict__ V, const int* __restrict__ mask,
                      bf16* __restrict__ attnOut) {
    __shared__ __align__(16) bf16 sQ [64 * 64];   // [row][d]   (A operand)
    __shared__ __align__(16) bf16 sK [64 * 64];   // [key][d]   (B operand for S)
    __shared__ __align__(16) bf16 sVt[64 * 64];   // [d][key]   (B operand for PV)
    __shared__ __align__(16) bf16 sP [4][16 * 64];// per-wave P [row][key] (A operand)
    __shared__ int sMaskI[64];

    int tid  = threadIdx.x;
    int lane = tid & 31, wave = tid >> 5;
    int half = lane >> 4, l = lane & 15;

    int qb = blockIdx.x * 64;
    int h  = blockIdx.y;
    int b  = blockIdx.z;
    size_t base = (size_t)(b * HEADS + h) * SEQ * HEAD_DIM;

    // load Q block: 64x64 bf16
    #pragma unroll
    for (int i = 0; i < 4; ++i) {
        int c = tid + 128 * i, row = c >> 3, kc = (c & 7) * 8;
        *(u32x4*)(sQ + row * 64 + kc) =
            *(const u32x4*)(Q + base + (size_t)(qb + row) * 64 + kc);
    }

    f32x8 o[4];
    float m_row[8], l_row[8];
    #pragma unroll
    for (int nt = 0; nt < 4; ++nt)
        #pragma unroll
        for (int e = 0; e < 8; ++e) o[nt][e] = 0.0f;
    #pragma unroll
    for (int e = 0; e < 8; ++e) { m_row[e] = NEG_INF; l_row[e] = 0.0f; }

    for (int kb = 0; kb < SEQ / 64; ++kb) {
        int kbase = kb * 64;
        // K block [key][d], V block transposed [d][key]
        #pragma unroll
        for (int i = 0; i < 4; ++i) {
            int c = tid + 128 * i, key = c >> 3, kc = (c & 7) * 8;
            *(u32x4*)(sK + key * 64 + kc) =
                *(const u32x4*)(Km + base + (size_t)(kbase + key) * 64 + kc);
            Chunk8 vc;
            vc.q = *(const u32x4*)(V + base + (size_t)(kbase + key) * 64 + kc);
            #pragma unroll
            for (int j = 0; j < 8; ++j)
                sVt[(kc + j) * 64 + key] = vc.h[j];
        }
        if (tid < 64) sMaskI[tid] = mask[b * SEQ + kbase + tid];
        __syncthreads();

        // S = Q K^T * SCALE (+mask): 16 rows x 64 keys per wave
        f32x8 s[4];
        #pragma unroll
        for (int ct = 0; ct < 4; ++ct) {
            f32x8 sc;
            #pragma unroll
            for (int e = 0; e < 8; ++e) sc[e] = 0.0f;
            #pragma unroll
            for (int kk = 0; kk < 2; ++kk) {
                bf16x16 af = load_frag_a(sQ, 64, wave * 16, kk * 32, lane);
                bf16x16 bfr = load_frag_b(sK, 64, ct * 16, kk * 32, lane);
                sc = wmma_bf16(af, bfr, sc);
            }
            int keep = sMaskI[ct * 16 + l];
            #pragma unroll
            for (int e = 0; e < 8; ++e)
                sc[e] = keep ? sc[e] * SCALE : NEG_INF;
            s[ct] = sc;
        }

        // online softmax: rows replicated across 16-lane halves
        #pragma unroll
        for (int e = 0; e < 8; ++e) {
            float mx = fmaxf(fmaxf(s[0][e], s[1][e]), fmaxf(s[2][e], s[3][e]));
            #pragma unroll
            for (int off = 1; off <= 8; off <<= 1)
                mx = fmaxf(mx, __shfl_xor(mx, off, 32));
            float m_new = fmaxf(m_row[e], mx);
            float alpha = __expf(m_row[e] - m_new);
            float rs = 0.0f;
            #pragma unroll
            for (int ct = 0; ct < 4; ++ct) {
                float p = __expf(s[ct][e] - m_new);
                s[ct][e] = p;
                rs += p;
            }
            #pragma unroll
            for (int off = 1; off <= 8; off <<= 1)
                rs += __shfl_xor(rs, off, 32);
            l_row[e] = l_row[e] * alpha + rs;
            m_row[e] = m_new;
            #pragma unroll
            for (int nt = 0; nt < 4; ++nt) o[nt][e] *= alpha;
        }

        // P (C layout) -> wave-private LDS [row][key] (A layout source)
        bf16* pw = &sP[wave][0];
        #pragma unroll
        for (int ct = 0; ct < 4; ++ct)
            #pragma unroll
            for (int e = 0; e < 8; ++e)
                pw[(e + half * 8) * 64 + ct * 16 + l] = f2bf16(s[ct][e]);

        // O += P * V
        #pragma unroll
        for (int kk = 0; kk < 2; ++kk) {
            bf16x16 af = load_frag_a(pw, 64, 0, kk * 32, lane);
            #pragma unroll
            for (int nt = 0; nt < 4; ++nt) {
                bf16x16 bfr = load_frag_b(sVt, 64, nt * 16, kk * 32, lane);
                o[nt] = wmma_bf16(af, bfr, o[nt]);
            }
        }
        __syncthreads();   // before next iteration overwrites sK/sVt
    }

    // normalize + write attnOut[b*N + row][h*64 + d] bf16
    #pragma unroll
    for (int e = 0; e < 8; ++e) {
        float inv = l_row[e] > 0.0f ? 1.0f / l_row[e] : 0.0f;
        int r = qb + wave * 16 + e + half * 8;
        size_t R = (size_t)(b * SEQ + r) * HIDDEN + h * HEAD_DIM;
        #pragma unroll
        for (int nt = 0; nt < 4; ++nt)
            attnOut[R + nt * 16 + l] = f2bf16(o[nt][e] * inv);
    }
}

// ---------------------------------------------------------------- launch

extern "C" void kernel_launch(void* const* d_in, const int* in_sizes, int n_in,
                              void* d_out, int out_size, void* d_ws, size_t ws_size,
                              hipStream_t stream) {
    const float* query = (const float*)d_in[0];
    const int*   mask  = (const int*)  d_in[1];
    const float* Wq = (const float*)d_in[2]; const float* bq = (const float*)d_in[3];
    const float* Wk = (const float*)d_in[4]; const float* bk = (const float*)d_in[5];
    const float* Wv = (const float*)d_in[6]; const float* bv = (const float*)d_in[7];
    const float* Wo = (const float*)d_in[8]; const float* bo = (const float*)d_in[9];

    char* ws = (char*)d_ws;
    const size_t MAT_BF16 = (size_t)ROWS * HIDDEN * sizeof(bf16);   // 16 MiB
    const size_t W_BF16   = (size_t)HIDDEN * HIDDEN * sizeof(bf16); //  2 MiB
    bf16* qryB  = (bf16*)ws;               ws += MAT_BF16;
    bf16* WqT   = (bf16*)ws;               ws += W_BF16;
    bf16* WkT   = (bf16*)ws;               ws += W_BF16;
    bf16* WvT   = (bf16*)ws;               ws += W_BF16;
    bf16* WoT   = (bf16*)ws;               ws += W_BF16;
    bf16* Qb    = (bf16*)ws;               ws += MAT_BF16;          // [B,H,N,D]
    bf16* Kb    = (bf16*)ws;               ws += MAT_BF16;
    bf16* Vb    = (bf16*)ws;               ws += MAT_BF16;
    bf16* attnO = (bf16*)ws;               ws += MAT_BF16;          // [B*N, H*D]

    // 1. query f32 -> bf16
    {
        int n4 = ROWS * HIDDEN / 4;
        convert_f32_to_bf16_kernel<<<n4 / 256, 256, 0, stream>>>(query, qryB, n4);
    }
    // 2. weight transpose+convert
    {
        dim3 g(HIDDEN / 32, HIDDEN / 32);
        transpose_convert_kernel<<<g, 256, 0, stream>>>(Wq, WqT);
        transpose_convert_kernel<<<g, 256, 0, stream>>>(Wk, WkT);
        transpose_convert_kernel<<<g, 256, 0, stream>>>(Wv, WvT);
        transpose_convert_kernel<<<g, 256, 0, stream>>>(Wo, WoT);
    }
    // 3. QKV projections -> [B,H,N,D] bf16
    {
        dim3 g(HIDDEN / 128, ROWS / 128);
        gemm_bf16_kernel<1><<<g, 256, 0, stream>>>(qryB, WqT, bq, (void*)Qb);
        gemm_bf16_kernel<1><<<g, 256, 0, stream>>>(qryB, WkT, bk, (void*)Kb);
        gemm_bf16_kernel<1><<<g, 256, 0, stream>>>(qryB, WvT, bv, (void*)Vb);
    }
    // 4. flash attention
    {
        dim3 g(SEQ / 64, HEADS, BATCH);
        attention_kernel<<<g, 128, 0, stream>>>(Qb, Kb, Vb, mask, attnO);
    }
    // 5. output projection -> f32 d_out
    {
        dim3 g(HIDDEN / 128, ROWS / 128);
        gemm_bf16_kernel<0><<<g, 256, 0, stream>>>(attnO, WoT, bo, d_out);
    }
}
